// AttentiveItemToVec_52072183496957
// MI455X (gfx1250) — matrix operand order repaired and checked by reading, not compiled, not run on hardware
//
#include <hip/hip_runtime.h>
#include <hip/hip_bf16.h>

// ---------- types ----------
typedef _Float16 half4  __attribute__((ext_vector_type(4)));
typedef _Float16 half8  __attribute__((ext_vector_type(8)));
typedef _Float16 half16 __attribute__((ext_vector_type(16)));
typedef float    float8 __attribute__((ext_vector_type(8)));

// D = A(16x32 f16) * B(32x16 f16) + C(16x16 f32)
#define WMMA_F16(A, B, C) \
    __builtin_amdgcn_wmma_f32_16x16x32_f16(false, (A), false, (B), (short)0, (C), false, false)

// ---------- problem constants ----------
#define EDIM  128
#define DA    60
#define DAP   64      // DA padded to 64
#define BT    8       // targets per batch
#define CC    200     // context items
#define CP16  208     // C padded to 13 tiles of 16
#define CP32  224     // C padded to 7 K-steps of 32
#define EPS   1e-6f

// ---------- WMMA fragment loads (ISA 7.12.2 layouts) ----------
__device__ __forceinline__ half16 load_a_frag(const _Float16* base, int pitch,
                                              int mbase, int kbase, int lane) {
    const _Float16* p = base + (mbase + (lane & 15)) * pitch + kbase + ((lane >> 4) * 8);
    half8 a0 = *(const half8*)(p);
    half8 a1 = *(const half8*)(p + 16);
    return __builtin_shufflevector(a0, a1, 0,1,2,3,4,5,6,7,8,9,10,11,12,13,14,15);
}

__device__ __forceinline__ half16 load_b_frag(const _Float16* baseT, int pitch,
                                              int nbase, int kbase, int lane) {
    const _Float16* p = baseT + (nbase + (lane & 15)) * pitch + kbase + ((lane >> 4) * 8);
    half8 b0 = *(const half8*)(p);
    half8 b1 = *(const half8*)(p + 16);
    return __builtin_shufflevector(b0, b1, 0,1,2,3,4,5,6,7,8,9,10,11,12,13,14,15);
}

// ---------- async global -> LDS (CDNA5 ASYNCcnt path) ----------
// per-lane: LDS[lds_off] = MEM[gaddr], 8 bytes; tracked by ASYNCcnt
__device__ __forceinline__ void async_copy_b64(unsigned lds_off, unsigned long long gaddr) {
    asm volatile("global_load_async_to_lds_b64 %0, %1, off"
                 :: "v"(lds_off), "v"(gaddr) : "memory");
}
__device__ __forceinline__ void wait_async0() {
    asm volatile("s_wait_asynccnt 0x0" ::: "memory");
}

// ---------- d_ws layout ----------
// f16 weight region (halves):
#define WS_ACT 0          // AcT [64][128] (rows >= 60 zero)
#define WS_ATT 8192       // AtT [64][128]
#define WS_BCT 16384      // BcT [128][128]
#define WS_RT  32768      // RT  [128][128]
#define WS_HALVES 49152
// f32 bias region (floats, starts at byte 98304):
#define WF_ACB 0
#define WF_ATB 64
#define WF_BCB 128
#define WF_RB  256
#define WF_FLOATS 384
// f16 embedding tables (bytes):
#define VOC 100000
#define WS2_CVEC  1048576ULL                               // [VOC][128] f16 = 25.6 MB
#define WS2_TVEC  (WS2_CVEC + (unsigned long long)VOC * EDIM * 2)
#define WS2_END   (WS2_TVEC + (unsigned long long)VOC * EDIM * 2)

__global__ __launch_bounds__(256)
void aitv_prep_weights(const float* __restrict__ Ac_w, const float* __restrict__ At_w,
                       const float* __restrict__ Bc_w, const float* __restrict__ R_w,
                       const float* __restrict__ Ac_b, const float* __restrict__ At_b,
                       const float* __restrict__ Bc_b, const float* __restrict__ R_b,
                       _Float16* __restrict__ ws) {
    int i = blockIdx.x * 256 + threadIdx.x;
    if (i < WS_HALVES) {
        float v;
        if (i < WS_ATT) {
            int j = i - WS_ACT, r = j >> 7, c = j & 127;
            v = (r < DA) ? Ac_w[r * EDIM + c] : 0.0f;
        } else if (i < WS_BCT) {
            int j = i - WS_ATT, r = j >> 7, c = j & 127;
            v = (r < DA) ? At_w[r * EDIM + c] : 0.0f;
        } else if (i < WS_RT) {
            v = Bc_w[i - WS_BCT];
        } else {
            v = R_w[i - WS_RT];
        }
        ws[i] = (_Float16)v;
    } else if (i < WS_HALVES + WF_FLOATS) {
        int j = i - WS_HALVES;
        float* wsf = (float*)(ws + WS_HALVES);
        float v;
        if (j < WF_ATB)       v = (j < DA) ? Ac_b[j] : 0.0f;
        else if (j < WF_BCB)  v = (j - WF_ATB < DA) ? At_b[j - WF_ATB] : 0.0f;
        else if (j < WF_RB)   v = Bc_b[j - WF_BCB];
        else                  v = R_b[j - WF_RB];
        wsf[j] = v;
    }
}

// convert the two embedding tables to f16 (vectorized: 4 floats per thread)
__global__ __launch_bounds__(256)
void aitv_prep_tables(const float* __restrict__ cvec_w, const float* __restrict__ tvec_w,
                      _Float16* __restrict__ cvec_h, _Float16* __restrict__ tvec_h) {
    int i = blockIdx.x * 256 + threadIdx.x;                // i indexes groups of 4
    const int N4 = VOC * EDIM / 4;                         // per table
    const float* src; _Float16* dst;
    if (i < N4)      { src = cvec_w; dst = cvec_h; }
    else if (i < 2 * N4) { i -= N4; src = tvec_w; dst = tvec_h; }
    else return;
    float4 v = *(const float4*)(src + (size_t)i * 4);
    half4 h = { (_Float16)v.x, (_Float16)v.y, (_Float16)v.z, (_Float16)v.w };
    *(half4*)(dst + (size_t)i * 4) = h;
}

// ---------- LDS layout (bytes; single static alloc => LDS base 0) ----------
#define L_UH    0                           // u f16 [208][128]        53248
#define L_BCUT  53248                       // bc_u^T f16 [128][224]   57344
#define L_CVH   110592                      // c_vecs f16 [208][64]    26624
#define L_TVH   137216                      // t_vecs f16 [16][64]      2048
#define L_VH    139264                      // v f16 [16][128]          4096
#define L_ATTN  143360                      // attn f16 [16][224]       7168
#define L_COS   150528                      // cos f32 [8][224] / alpha f16 [16][128]
#define L_CN    157696                      // inv-cn f32 [208] (pad=1) 832
#define L_TN    158528                      // inv-tn f32 [8]            32
#define L_TOTAL 158560

template <bool F16TAB>
__global__ __launch_bounds__(256)
void aitv_main(const float* __restrict__ tvec_w, const float* __restrict__ cvec_w,
               const _Float16* __restrict__ tvec_h, const _Float16* __restrict__ cvec_h,
               const int* __restrict__ titems, const int* __restrict__ citems,
               const unsigned char* __restrict__ mask,
               const _Float16* __restrict__ ws, float* __restrict__ out) {
    __shared__ __align__(32) unsigned char smem[L_TOTAL];
    _Float16* uh    = (_Float16*)(smem + L_UH);
    _Float16* bcuT  = (_Float16*)(smem + L_BCUT);
    _Float16* cvh   = (_Float16*)(smem + L_CVH);
    _Float16* tvh   = (_Float16*)(smem + L_TVH);
    _Float16* vh    = (_Float16*)(smem + L_VH);
    _Float16* attn  = (_Float16*)(smem + L_ATTN);
    float*    cosb  = (float*)   (smem + L_COS);
    _Float16* alpha = (_Float16*)(smem + L_COS);   // reuse after softmax
    float*    cn    = (float*)   (smem + L_CN);    // 1/norm, padded with 1.0
    float*    tn    = (float*)   (smem + L_TN);    // 1/norm

    const int tid  = threadIdx.x;
    const int lane = tid & 31;
    const int wid  = __builtin_amdgcn_readfirstlane(tid >> 5);
    const int b    = blockIdx.x;
    const int hbase = (lane < 16) ? 0 : 8;

    const _Float16* AcT = ws + WS_ACT;
    const _Float16* AtT = ws + WS_ATT;
    const _Float16* BcT = ws + WS_BCT;
    const _Float16* RT  = ws + WS_RT;
    const float* wsf  = (const float*)(ws + WS_HALVES);
    const float* AcbP = wsf + WF_ACB;
    const float* AtbP = wsf + WF_ATB;
    const float* BcbP = wsf + WF_BCB;
    const float* RbP  = wsf + WF_RB;

    // ---- phase 0: gather embeddings -> LDS f16 ----
    if constexpr (F16TAB) {
        // pure DMA: per-lane 8B async copies of f16 rows, no VALU conversion
        for (int c = wid; c < CC; c += 8) {
            int idx = citems[b * CC + c];
            async_copy_b64((unsigned)(L_UH + c * 256 + lane * 8),
                           (unsigned long long)(uintptr_t)cvec_h +
                               (unsigned long long)idx * 256ULL + (unsigned)(lane * 8));
        }
        if (wid < BT) {
            int idx = titems[b * BT + wid];
            async_copy_b64((unsigned)(L_VH + wid * 256 + lane * 8),
                           (unsigned long long)(uintptr_t)tvec_h +
                               (unsigned long long)idx * 256ULL + (unsigned)(lane * 8));
        }
        // zero pad rows (uh 200..207, vh 8..15): 256 threads x half4 each
        *(half4*)(uh + (CC + (tid >> 5)) * EDIM + (tid & 31) * 4) = half4{};
        *(half4*)(vh + (BT + (tid >> 5)) * EDIM + (tid & 31) * 4) = half4{};
    } else {
        for (int c = wid; c < CP16; c += 8) {
            float4 v = make_float4(0.f, 0.f, 0.f, 0.f);
            if (c < CC) {
                int idx = citems[b * CC + c];
                v = *(const float4*)(cvec_w + (size_t)idx * EDIM + lane * 4);
            }
            half4 h = { (_Float16)v.x, (_Float16)v.y, (_Float16)v.z, (_Float16)v.w };
            *(half4*)(uh + c * EDIM + lane * 4) = h;
        }
        for (int t = wid; t < 16; t += 8) {
            float4 v = make_float4(0.f, 0.f, 0.f, 0.f);
            if (t < BT) {
                int idx = titems[b * BT + t];
                v = *(const float4*)(tvec_w + (size_t)idx * EDIM + lane * 4);
            }
            half4 h = { (_Float16)v.x, (_Float16)v.y, (_Float16)v.z, (_Float16)v.w };
            *(half4*)(vh + t * EDIM + lane * 4) = h;
        }
    }
    for (int i = tid; i < EDIM * 16; i += 256)             // bcuT cols 208..223 = 0
        bcuT[(i >> 4) * CP32 + CP16 + (i & 15)] = (_Float16)0.f;
    for (int i = tid; i < 8 * CP32; i += 256)              // attn rows 8..15 = 0
        attn[(8 + i / CP32) * CP32 + (i % CP32)] = (_Float16)0.f;
    if constexpr (F16TAB) wait_async0();                   // own DMAs done, then barrier
    __syncthreads();

    // ---- phase 1 (flipped): c_vecsT = Ac @ u^T -> cvh[c][d] ; t_vecsT -> tvh[t][d] ----
    for (int job = wid; job < 56; job += 8) {
        if (job < 52) {
            int mt = job / 13, nt = job - mt * 13;
            float8 acc = {};
            for (int kk = 0; kk < 4; ++kk) {
                half16 a = load_a_frag(AcT, EDIM, mt * 16, kk * 32, lane);
                half16 bm = load_b_frag(uh, EDIM, nt * 16, kk * 32, lane);
                acc = WMMA_F16(a, bm, acc);
            }
            int c = nt * 16 + (lane & 15);
            int dbase = mt * 16 + hbase;
            float8 bias = *(const float8*)(AcbP + dbase);
            half8 h;
            for (int r = 0; r < 8; ++r) h[r] = (_Float16)(acc[r] + bias[r]);
            *(half8*)(cvh + c * DAP + dbase) = h;
        } else {
            int mt = job - 52;
            float8 acc = {};
            for (int kk = 0; kk < 4; ++kk) {
                half16 a = load_a_frag(AtT, EDIM, mt * 16, kk * 32, lane);
                half16 bm = load_b_frag(vh, EDIM, 0, kk * 32, lane);
                acc = WMMA_F16(a, bm, acc);
            }
            int t = lane & 15;
            int dbase = mt * 16 + hbase;
            float8 bias = *(const float8*)(AtbP + dbase);
            half8 h;
            for (int r = 0; r < 8; ++r) h[r] = (_Float16)(acc[r] + bias[r]);
            *(half8*)(tvh + t * DAP + dbase) = h;
        }
    }
    __syncthreads();

    // ---- phase 1b: inverse eps-clamped norms (cn padded with 1.0) ----
    if (tid < CC) {
        float s = 0.f;
        for (int d = 0; d < DAP; ++d) { float x = (float)cvh[tid * DAP + d]; s += x * x; }
        cn[tid] = 1.0f / fmaxf(sqrtf(s), EPS);
    } else if (tid < CP16) {
        cn[tid] = 1.0f;
    } else if (tid < CP16 + BT) {
        int t = tid - CP16;
        float s = 0.f;
        for (int d = 0; d < DAP; ++d) { float x = (float)tvh[t * DAP + d]; s += x * x; }
        tn[t] = 1.0f / fmaxf(sqrtf(s), EPS);
    }
    __syncthreads();

    // ---- phase 2+3: bc_u^T -> bcuT[e][c] ; cosT -> cosb[t][c] ----
    for (int job = wid; job < 117; job += 8) {
        if (job < 104) {
            int mt = job >> 3, nt = job & 7;
            float8 acc = {};
            for (int kk = 0; kk < 4; ++kk) {
                half16 a = load_a_frag(uh, EDIM, mt * 16, kk * 32, lane);
                half16 bm = load_b_frag(BcT, EDIM, nt * 16, kk * 32, lane);
                acc = WMMA_F16(a, bm, acc);
            }
            int e = nt * 16 + (lane & 15);
            float bias = BcbP[e];
            int cbase = mt * 16 + hbase;
            half8 h;
            for (int r = 0; r < 8; ++r) h[r] = (_Float16)(acc[r] + bias);
            *(half8*)(bcuT + e * CP32 + cbase) = h;
        } else {
            int mt = job - 104;
            float8 acc = {};
            for (int kk = 0; kk < 2; ++kk) {
                half16 a = load_a_frag(cvh, DAP, mt * 16, kk * 32, lane);
                half16 bm = load_b_frag(tvh, DAP, 0, kk * 32, lane);
                acc = WMMA_F16(a, bm, acc);
            }
            int t = lane & 15;
            if (t < BT) {
                int cbase = mt * 16 + hbase;
                float tni = tn[t];
                float8 cni = *(const float8*)(cn + cbase);
                float8 v;
                for (int r = 0; r < 8; ++r) v[r] = acc[r] * tni * cni[r];
                *(float8*)(cosb + t * CP32 + cbase) = v;
            }
        }
    }
    __syncthreads();

    // ---- phase 4: masked softmax over C (wave w -> row t=w) ----
    {
        int t = wid;
        const unsigned char* mrow = mask + (size_t)b * CC;
        float vals[7], mx = -__builtin_inff();
        for (int i = 0; i < 7; ++i) {
            int c = lane + i * 32;
            float v = -__builtin_inff();
            if (c < CC && !mrow[c]) v = cosb[t * CP32 + c];
            vals[i] = v;
            mx = fmaxf(mx, v);
        }
        for (int off = 16; off > 0; off >>= 1) mx = fmaxf(mx, __shfl_xor(mx, off, 32));
        float e[7], sum = 0.f;
        for (int i = 0; i < 7; ++i) {
            e[i] = (vals[i] == -__builtin_inff()) ? 0.f : __expf(vals[i] - mx);
            sum += e[i];
        }
        for (int off = 16; off > 0; off >>= 1) sum += __shfl_xor(sum, off, 32);
        float inv = 1.0f / sum;
        for (int i = 0; i < 7; ++i)
            attn[t * CP32 + lane + i * 32] = (_Float16)(e[i] * inv);
    }
    __syncthreads();

    // ---- phase 5 (flipped): alphaT = bc_u^T @ attn^T -> alpha[t][e] ----
    {
        int mt = wid;
        float8 acc = {};
        for (int kk = 0; kk < 7; ++kk) {
            half16 a = load_a_frag(bcuT, CP32, mt * 16, kk * 32, lane);
            half16 bm = load_b_frag(attn, CP32, 0, kk * 32, lane);
            acc = WMMA_F16(a, bm, acc);
        }
        int t = lane & 15;
        int ebase = mt * 16 + hbase;
        half8 h;
        for (int r = 0; r < 8; ++r) h[r] = (_Float16)acc[r];
        *(half8*)(alpha + t * EDIM + ebase) = h;
    }
    __syncthreads();

    // ---- phase 6 (flipped): zT = R @ alpha^T (+b) -> global ----
    {
        int mt = wid;
        float8 acc = {};
        for (int kk = 0; kk < 4; ++kk) {
            half16 a = load_a_frag(RT, EDIM, mt * 16, kk * 32, lane);
            half16 bm = load_b_frag(alpha, EDIM, 0, kk * 32, lane);
            acc = WMMA_F16(a, bm, acc);
        }
        int t = lane & 15;
        if (t < BT) {
            int ebase = mt * 16 + hbase;
            float8 bias = *(const float8*)(RbP + ebase);
            float* o = out + ((size_t)b * BT + t) * EDIM + ebase;
            float8 v;
            for (int r = 0; r < 8; ++r) v[r] = acc[r] + bias[r];
            *(float8*)o = v;
        }
    }
}

extern "C" void kernel_launch(void* const* d_in, const int* in_sizes, int n_in,
                              void* d_out, int out_size, void* d_ws, size_t ws_size,
                              hipStream_t stream) {
    const float* tvec_w = (const float*)d_in[0];
    const float* cvec_w = (const float*)d_in[1];
    const float* Ac_w   = (const float*)d_in[2];
    const float* Ac_b   = (const float*)d_in[3];
    const float* At_w   = (const float*)d_in[4];
    const float* At_b   = (const float*)d_in[5];
    const float* Bc_w   = (const float*)d_in[6];
    const float* Bc_b   = (const float*)d_in[7];
    const float* R_w    = (const float*)d_in[8];
    const float* R_b    = (const float*)d_in[9];
    const int*   titems = (const int*)d_in[10];
    const int*   citems = (const int*)d_in[11];
    const unsigned char* mask = (const unsigned char*)d_in[12];
    _Float16* ws = (_Float16*)d_ws;
    float* out = (float*)d_out;

    int nB = in_sizes[10] / BT;                            // 4096

    aitv_prep_weights<<<(WS_HALVES + WF_FLOATS + 255) / 256, 256, 0, stream>>>(
        Ac_w, At_w, Bc_w, R_w, Ac_b, At_b, Bc_b, R_b, ws);

    if (ws_size >= (size_t)WS2_END) {
        // f16 embedding tables fit in scratch: convert once, then async-DMA gather
        _Float16* cvec_h = (_Float16*)((char*)d_ws + WS2_CVEC);
        _Float16* tvec_h = (_Float16*)((char*)d_ws + WS2_TVEC);
        int n4 = 2 * (VOC * EDIM / 4);
        aitv_prep_tables<<<(n4 + 255) / 256, 256, 0, stream>>>(cvec_w, tvec_w, cvec_h, tvec_h);
        aitv_main<true><<<nB, 256, 0, stream>>>(tvec_w, cvec_w, tvec_h, cvec_h,
                                                titems, citems, mask, ws, out);
    } else {
        aitv_main<false><<<nB, 256, 0, stream>>>(tvec_w, cvec_w, nullptr, nullptr,
                                                 titems, citems, mask, ws, out);
    }
}
